// SinkhornDistance_47485158425241
// MI455X (gfx1250) — matrix operand order repaired and checked by reading, not compiled, not run on hardware
//
#include <hip/hip_runtime.h>
#include <hip/hip_bf16.h>

// Problem constants (from reference setup)
#define BATCH 256
#define DIMD  128
#define NQ    512
#define NS    512
#define REGI  10.0f      // 1/REG
#define EPSV  1e-8f

typedef __attribute__((ext_vector_type(16))) _Float16 v16h;
typedef __attribute__((ext_vector_type(8)))  float    v8f;

union FragAB { uint4 u[2]; v16h h; };
union Half8  { uint4 u; _Float16 h[8]; };

// ---------------------------------------------------------------------------
// CDNA5 async global->LDS copy (16B per lane), tracked by ASYNCcnt.
// LDS address = low 32 bits of the flat shared pointer (LDS aperture).
// ---------------------------------------------------------------------------
__device__ __forceinline__ void async_copy_b128(const _Float16* gsrc,
                                                _Float16* ldst) {
    unsigned lds = (unsigned)(uintptr_t)ldst;
    unsigned long long ga = (unsigned long long)(uintptr_t)gsrc;
    asm volatile("global_load_async_to_lds_b128 %0, %1, off"
                 :: "v"(lds), "v"(ga) : "memory");
}

__device__ __forceinline__ void wait_async0() {
#if __has_builtin(__builtin_amdgcn_s_wait_asynccnt)
    __builtin_amdgcn_s_wait_asynccnt(0);
#else
    asm volatile("s_wait_asynccnt 0x0" ::: "memory");
#endif
}

// ---------------------------------------------------------------------------
// Init: out = 0 (u/v live in LDS inside the fused iteration kernel)
// ---------------------------------------------------------------------------
__global__ void sk_init_kernel(float* __restrict__ out) {
    int idx = blockIdx.x * blockDim.x + threadIdx.x;
    if (idx < BATCH) out[idx] = 0.0f;
}

// ---------------------------------------------------------------------------
// Tiled transpose + f32->f16: in (B, R, C) -> out (B, C, R)
// R = D = 128, C = 512.  block (32,8), grid (C/32, R/32, B)
// ---------------------------------------------------------------------------
__global__ void sk_transpose_f16_kernel(const float* __restrict__ in,
                                        _Float16* __restrict__ out) {
    __shared__ float tile[32][33];
    const int R = DIMD, C = NQ; // NQ == NS == 512
    int tx = threadIdx.x, ty = threadIdx.y;
    int c0 = blockIdx.x * 32, r0 = blockIdx.y * 32, b = blockIdx.z;
    const float* inb = in + (size_t)b * R * C;
    _Float16* outb = out + (size_t)b * C * R;
#pragma unroll
    for (int j = ty; j < 32; j += 8)
        tile[j][tx] = inb[(size_t)(r0 + j) * C + c0 + tx];
    __syncthreads();
#pragma unroll
    for (int j = ty; j < 32; j += 8)
        outb[(size_t)(c0 + j) * R + r0 + tx] = (_Float16)tile[tx][j];
}

// ---------------------------------------------------------------------------
// Batched GEMM sim = qT * sT^T  (M=NQ rows, N=NS cols, K=D) with WMMA f16,
// then K = exp((sim-1)*10) stored as f16 (134 MB total -> L2-resident).
// Tiles staged with async global->LDS (no VGPR round trip).
// grid (16, B): blockIdx.x -> 4x4 tiles of 128x128. block = 256 (8 waves).
// ---------------------------------------------------------------------------
__global__ void sk_gemm_exp_kernel(const _Float16* __restrict__ qT,  // (B,NQ,D)
                                   const _Float16* __restrict__ sT,  // (B,NS,D)
                                   _Float16* __restrict__ Kmat) {    // (B,NQ,NS)
    __shared__ _Float16 At[128 * 32];
    __shared__ _Float16 Bt[128 * 32];

    const int tid = threadIdx.x;
    const int w = tid >> 5, l = tid & 31;
    const int lr = l & 15;        // row/col within 16x16 subtile
    const int hi = l >> 4;        // lane half select

    const int tilei = blockIdx.x;
    const int m0 = (tilei >> 2) * 128;
    const int n0 = (tilei & 3) * 128;
    const int b = blockIdx.y;

    const _Float16* qb = qT + (size_t)b * NQ * DIMD;
    const _Float16* sb = sT + (size_t)b * NS * DIMD;

    v8f acc[8];
#pragma unroll
    for (int nn = 0; nn < 8; ++nn) acc[nn] = (v8f){0,0,0,0,0,0,0,0};

    for (int kc = 0; kc < DIMD; kc += 32) {
        __syncthreads();
#pragma unroll
        for (int i = tid; i < 512; i += 256) {
            int row = i >> 2, seg = i & 3;
            async_copy_b128(qb + (size_t)(m0 + row) * DIMD + kc + seg * 8,
                            &At[row * 32 + seg * 8]);
            async_copy_b128(sb + (size_t)(n0 + row) * DIMD + kc + seg * 8,
                            &Bt[row * 32 + seg * 8]);
        }
        wait_async0();
        __syncthreads();

        FragAB a;
        int aoff = (w * 16 + lr) * 32 + hi * 8;
        a.u[0] = *(const uint4*)&At[aoff];
        a.u[1] = *(const uint4*)&At[aoff + 16];
#pragma unroll
        for (int nn = 0; nn < 8; ++nn) {
            FragAB bf;
            int boff = (nn * 16 + lr) * 32 + hi * 8;
            bf.u[0] = *(const uint4*)&Bt[boff];
            bf.u[1] = *(const uint4*)&Bt[boff + 16];
            acc[nn] = __builtin_amdgcn_wmma_f32_16x16x32_f16(
                false, a.h, false, bf.h, (short)0, acc[nn], false, false);
        }
    }

    _Float16* Kb = Kmat + (size_t)b * NQ * NS;
#pragma unroll
    for (int nn = 0; nn < 8; ++nn) {
#pragma unroll
        for (int j = 0; j < 8; ++j) {
            int row_g = m0 + w * 16 + j + hi * 8;
            int col_g = n0 + nn * 16 + lr;
            float sim = acc[nn][j];
            Kb[(size_t)row_g * NS + col_g] = (_Float16)__expf((sim - 1.0f) * REGI);
        }
    }
}

// ---------------------------------------------------------------------------
// Fused Sinkhorn iterations: one block (512 threads) per batch, u/v in LDS.
//   repeat 10x:  v[m] = nu / (sum_n K[n,m]*u[n] + eps)   (col phase)
//                u[n] = mu / (sum_m K[n,m]*v[m] + eps)   (row phase)
// Col phase: 8 groups x 64 threads, each thread owns 8 consecutive columns
// (b128 K loads + WGP-level prefetch of the next row), partials merged with
// LDS atomic ds_add_f32. Row phase: wave per row, b128 loads, shfl reduce.
// ---------------------------------------------------------------------------
__global__ void sk_sinkhorn_kernel(const _Float16* __restrict__ Kmat,
                                   float* __restrict__ u_out,
                                   float* __restrict__ v_out) {
    __shared__ float us[NQ];
    __shared__ float vs[NS];

    const int b = blockIdx.x, tid = threadIdx.x;
    const _Float16* Kb = Kmat + (size_t)b * NQ * NS;

    const int g = tid >> 6;        // col-phase group 0..7
    const int t = tid & 63;        // thread in group -> columns t*8..t*8+7
    const int w = tid >> 5;        // row-phase wave 0..15
    const int l = tid & 31;

    const float nu = 1.0f / (float)NS;
    const float mu = 1.0f / (float)NQ;

    us[tid] = 1.0f;                // u0 = ones
    __syncthreads();

    for (int it = 0; it < 10; ++it) {
        // ---- column phase: vs = nu / (K^T u + eps) ----
        vs[tid] = 0.0f;
        __syncthreads();
        {
            float acc[8] = {0.f, 0.f, 0.f, 0.f, 0.f, 0.f, 0.f, 0.f};
            for (int n = g; n < NQ; n += 8) {
                __builtin_prefetch(Kb + (size_t)(n + 8) * NS + t * 8, 0, 3);
                float un = us[n];
                Half8 kv;
                kv.u = *(const uint4*)(Kb + (size_t)n * NS + t * 8);
#pragma unroll
                for (int e = 0; e < 8; ++e)
                    acc[e] += (float)kv.h[e] * un;
            }
#pragma unroll
            for (int e = 0; e < 8; ++e)
                atomicAdd(&vs[t * 8 + e], acc[e]);
        }
        __syncthreads();
        vs[tid] = nu / (vs[tid] + EPSV);
        __syncthreads();

        // ---- row phase: us = mu / (K v + eps) ----
        for (int n = w; n < NQ; n += 16) {
            float sum = 0.f;
#pragma unroll
            for (int j = 0; j < 2; ++j) {
                int m = l * 8 + j * 256;
                Half8 kv;
                kv.u = *(const uint4*)(Kb + (size_t)n * NS + m);
#pragma unroll
                for (int e = 0; e < 8; ++e)
                    sum += (float)kv.h[e] * vs[m + e];
            }
#pragma unroll
            for (int off = 16; off > 0; off >>= 1)
                sum += __shfl_xor(sum, off, 32);
            if (l == 0) us[n] = mu / (sum + EPSV);
        }
        __syncthreads();
    }

    u_out[(size_t)b * NQ + tid] = us[tid];
    v_out[(size_t)b * NS + tid] = vs[tid];
}

// ---------------------------------------------------------------------------
// distance[b] += sum_{n,m in tile} u_n * K(sim) * v_m * (1 - sim)
// sim recomputed by WMMA (operands are L2-resident), K regenerated via exp.
// ---------------------------------------------------------------------------
__global__ void sk_final_kernel(const _Float16* __restrict__ qT,
                                const _Float16* __restrict__ sT,
                                const float* __restrict__ u,
                                const float* __restrict__ v,
                                float* __restrict__ out) {
    __shared__ _Float16 At[128 * 32];
    __shared__ _Float16 Bt[128 * 32];
    __shared__ float useg[128];
    __shared__ float vseg[128];
    __shared__ float wred[8];

    const int tid = threadIdx.x;
    const int w = tid >> 5, l = tid & 31;
    const int lr = l & 15;
    const int hi = l >> 4;

    const int tilei = blockIdx.x;
    const int m0 = (tilei >> 2) * 128;
    const int n0 = (tilei & 3) * 128;
    const int b = blockIdx.y;

    if (tid < 128) useg[tid] = u[(size_t)b * NQ + m0 + tid];
    else           vseg[tid - 128] = v[(size_t)b * NS + n0 + (tid - 128)];

    const _Float16* qb = qT + (size_t)b * NQ * DIMD;
    const _Float16* sb = sT + (size_t)b * NS * DIMD;

    v8f acc[8];
#pragma unroll
    for (int nn = 0; nn < 8; ++nn) acc[nn] = (v8f){0,0,0,0,0,0,0,0};

    for (int kc = 0; kc < DIMD; kc += 32) {
        __syncthreads();
#pragma unroll
        for (int i = tid; i < 512; i += 256) {
            int row = i >> 2, seg = i & 3;
            async_copy_b128(qb + (size_t)(m0 + row) * DIMD + kc + seg * 8,
                            &At[row * 32 + seg * 8]);
            async_copy_b128(sb + (size_t)(n0 + row) * DIMD + kc + seg * 8,
                            &Bt[row * 32 + seg * 8]);
        }
        wait_async0();
        __syncthreads();

        FragAB a;
        int aoff = (w * 16 + lr) * 32 + hi * 8;
        a.u[0] = *(const uint4*)&At[aoff];
        a.u[1] = *(const uint4*)&At[aoff + 16];
#pragma unroll
        for (int nn = 0; nn < 8; ++nn) {
            FragAB bf;
            int boff = (nn * 16 + lr) * 32 + hi * 8;
            bf.u[0] = *(const uint4*)&Bt[boff];
            bf.u[1] = *(const uint4*)&Bt[boff + 16];
            acc[nn] = __builtin_amdgcn_wmma_f32_16x16x32_f16(
                false, a.h, false, bf.h, (short)0, acc[nn], false, false);
        }
    }

    float local = 0.f;
#pragma unroll
    for (int nn = 0; nn < 8; ++nn) {
#pragma unroll
        for (int j = 0; j < 8; ++j) {
            int rloc = w * 16 + j + hi * 8;
            int cloc = nn * 16 + lr;
            float sim = acc[nn][j];
            float Kv = __expf((sim - 1.0f) * REGI);
            local += useg[rloc] * Kv * vseg[cloc] * (1.0f - sim);
        }
    }
#pragma unroll
    for (int off = 16; off > 0; off >>= 1)
        local += __shfl_xor(local, off, 32);
    if (l == 0) wred[w] = local;
    __syncthreads();
    if (tid == 0) {
        float s = 0.f;
#pragma unroll
        for (int i = 0; i < 8; ++i) s += wred[i];
        atomicAdd(&out[b], s);
    }
}

// ---------------------------------------------------------------------------
// Launch
// ---------------------------------------------------------------------------
extern "C" void kernel_launch(void* const* d_in, const int* in_sizes, int n_in,
                              void* d_out, int out_size, void* d_ws, size_t ws_size,
                              hipStream_t stream) {
    const float* q = (const float*)d_in[0];   // (B, D, NQ)
    const float* s = (const float*)d_in[1];   // (B, D, NS)
    float* out = (float*)d_out;               // (B,)

    char* ws = (char*)d_ws;
    constexpr size_t QT_OFF = 0;
    constexpr size_t QT_SZ  = (size_t)BATCH * NQ * DIMD * sizeof(_Float16); // 32 MB
    constexpr size_t ST_OFF = QT_OFF + QT_SZ;
    constexpr size_t ST_SZ  = (size_t)BATCH * NS * DIMD * sizeof(_Float16); // 32 MB
    constexpr size_t K_OFF  = ST_OFF + ST_SZ;
    constexpr size_t K_SZ   = (size_t)BATCH * NQ * NS * sizeof(_Float16);   // 128 MB (L2-resident)
    constexpr size_t U_OFF  = K_OFF + K_SZ;
    constexpr size_t U_SZ   = (size_t)BATCH * NQ * sizeof(float);
    constexpr size_t V_OFF  = U_OFF + U_SZ;

    _Float16* qT   = (_Float16*)(ws + QT_OFF);
    _Float16* sT   = (_Float16*)(ws + ST_OFF);
    _Float16* Kmat = (_Float16*)(ws + K_OFF);
    float*    u    = (float*)(ws + U_OFF);
    float*    v    = (float*)(ws + V_OFF);

    // out = 0
    sk_init_kernel<<<1, 256, 0, stream>>>(out);

    // f32 (B,D,N) -> f16 (B,N,D), K-contiguous for WMMA fragment loads
    dim3 tgrid(NQ / 32, DIMD / 32, BATCH);
    dim3 tblk(32, 8);
    sk_transpose_f16_kernel<<<tgrid, tblk, 0, stream>>>(q, qT);
    sk_transpose_f16_kernel<<<tgrid, tblk, 0, stream>>>(s, sT);

    // sim = qT * sT^T via WMMA, K = exp((sim-1)/REG) stored f16 (fits in L2)
    dim3 ggrid(16, BATCH);
    sk_gemm_exp_kernel<<<ggrid, 256, 0, stream>>>(qT, sT, Kmat);

    // all 10 Sinkhorn iterations fused: one block per batch, u/v in LDS
    sk_sinkhorn_kernel<<<BATCH, 512, 0, stream>>>(Kmat, u, v);

    // distance = <diag(u) K diag(v), cost>, sim recomputed with WMMA
    sk_final_kernel<<<ggrid, 256, 0, stream>>>(qT, sT, u, v, out);

    (void)in_sizes; (void)n_in; (void)out_size; (void)ws_size;
}
